// ScaledDotProductAttention_4982162063737
// MI455X (gfx1250) — matrix operand order repaired and checked by reading, not compiled, not run on hardware
//
#include <hip/hip_runtime.h>

// ---------------------------------------------------------------------------
// Scaled dot-product attention (B=2,H=8,S=2048,D=64) for gfx1250 (MI455X).
// f16 WMMA (v_wmma_f32_16x16x32_f16) with f32 accumulate; fp32 softmax.
// Outputs: O (B,H,S,D) then attn_dist (B,H,S,S), concatenated in d_out.
// ---------------------------------------------------------------------------

namespace {
constexpr int kS  = 2048;
constexpr int kD  = 64;
constexpr int kBH = 16;            // B*H
constexpr float kNegInf = 1.0e9f;

typedef __attribute__((ext_vector_type(16))) _Float16 v16h;
typedef __attribute__((ext_vector_type(8)))  _Float16 v8h;
typedef __attribute__((ext_vector_type(8)))  float    v8f;

__device__ inline v16h join8(v8h lo, v8h hi) {
  v16h r;
#pragma unroll
  for (int i = 0; i < 8; ++i) { r[i] = lo[i]; r[i + 8] = hi[i]; }
  return r;
}

__device__ inline v8f wmma_f16(v16h a, v16h b, v8f c) {
  // (neg_a, A, neg_b, B, c_mod, C, reuse_a, reuse_b)
  return __builtin_amdgcn_wmma_f32_16x16x32_f16(false, a, false, b,
                                                (short)0, c, false, false);
}
}  // namespace

// Pre-pass: K -> f16 row-major, V -> f16 transposed [bh][d][s].
__global__ void convert_kv(const float* __restrict__ K,
                           const float* __restrict__ V,
                           _Float16* __restrict__ Kh,
                           _Float16* __restrict__ Vt) {
  int idx = blockIdx.x * blockDim.x + threadIdx.x;   // 0 .. kBH*kS*kD-1
  Kh[idx] = (_Float16)K[idx];
  // Interpret idx as (bh, d, s) for the transposed V (coalesced Vt writes).
  int bh  = idx >> 17;                 // / (kD*kS)
  int rem = idx & (kD * kS - 1);
  int d   = rem >> 11;                 // / kS
  int s   = rem & (kS - 1);
  Vt[idx] = (_Float16)V[((size_t)(bh * kS) + s) * kD + d];
}

__global__ __launch_bounds__(128) void attn_kernel(
    const float* __restrict__ Q,
    const _Float16* __restrict__ Kh,
    const _Float16* __restrict__ Vt,
    const unsigned char* __restrict__ mask,
    float* __restrict__ outO,
    float* __restrict__ outA) {
  // Per-wave 16x32 P staging tile (f16), stride 40 keeps rows 16B-aligned.
  __shared__ _Float16 pstage[4][16][40];

  const int tid  = threadIdx.x;
  const int wave = tid >> 5;
  const int lane = tid & 31;
  const int ln   = lane & 15;   // col within tile (B/C), row within tile (A)
  const int g    = lane >> 4;   // lane group (0: rows 0-7 of C, 1: rows 8-15)

  const int bh = blockIdx.y;
  const int b  = bh >> 3;       // H == 8
  const int q0 = blockIdx.x * 64 + wave * 16;

  // ---- Q A-fragments: load f32, pre-scale by 1/sqrt(D)=0.125, cvt f16 ----
  // A 16x32 layout: lane (row=ln, group g), element e -> K = g*16 + e
  const float* qrow = Q + ((size_t)bh * kS + q0 + ln) * kD;
  v16h aQ[2];
#pragma unroll
  for (int t = 0; t < 2; ++t) {
    const int d0 = t * 32;
    v16h a;
#pragma unroll
    for (int e = 0; e < 8; ++e) {
      a[e]     = (_Float16)(qrow[d0 + g * 16 + e]      * 0.125f);
      a[e + 8] = (_Float16)(qrow[d0 + g * 16 + 8 + e] * 0.125f);
    }
    aQ[t] = a;
  }

  const _Float16* kbase = Kh + (size_t)bh * kS * kD;
  const unsigned char* mrow = mask + (size_t)b * kS;

  // Score tile for keys [n0, n0+16): C[m,n] = sum_d Qs[m,d]*K[n0+n,d], masked.
  auto score_tile = [&](int n0) -> v8f {
    const _Float16* krow = kbase + (size_t)(n0 + ln) * kD;
    v8f s = {};
    // B 32x16 layout: lane (col=ln, group g), element e -> K(d) = g*16 + e
    s = wmma_f16(aQ[0], *(const v16h*)(krow + g * 16), s);
    s = wmma_f16(aQ[1], *(const v16h*)(krow + 32 + g * 16), s);
    if (mrow[n0 + ln]) {
#pragma unroll
      for (int r = 0; r < 8; ++r) s[r] = -kNegInf;
    }
    return s;
  };

  // ---- Pass A: online (rowmax, rowsum) over all 2048 keys ----
  float rm[8], rl[8];
#pragma unroll
  for (int r = 0; r < 8; ++r) { rm[r] = -INFINITY; rl[r] = 0.f; }

  for (int n0 = 0; n0 < kS; n0 += 16) {
    v8f s = score_tile(n0);
#pragma unroll
    for (int r = 0; r < 8; ++r) {
      float nm = fmaxf(rm[r], s[r]);
      rl[r] = rl[r] * __expf(rm[r] - nm) + __expf(s[r] - nm);
      rm[r] = nm;
    }
  }

  // Cross-lane reduce within each 16-lane group (rows of this group).
  float rinv[8];
#pragma unroll
  for (int r = 0; r < 8; ++r) {
    float m = rm[r], l = rl[r];
#pragma unroll
    for (int off = 1; off < 16; off <<= 1) {
      float mo = __shfl_xor(m, off, 32);
      float lo = __shfl_xor(l, off, 32);
      float nm = fmaxf(m, mo);
      l = l * __expf(m - nm) + lo * __expf(mo - nm);
      m = nm;
    }
    rm[r]   = m;
    rinv[r] = 1.0f / l;
  }

  // ---- Pass B: recompute scores, emit attn_dist, accumulate O = P*V ----
  v8f acc[4];
#pragma unroll
  for (int dt = 0; dt < 4; ++dt) acc[dt] = (v8f){};

  float* arow = outA + ((size_t)bh * kS + q0) * kS;
  const _Float16* vtb = Vt + (size_t)bh * kD * kS;

  for (int k0 = 0; k0 < kS; k0 += 32) {
#pragma unroll
    for (int t = 0; t < 2; ++t) {
      const int n0 = k0 + t * 16;
      v8f s = score_tile(n0);
#pragma unroll
      for (int r = 0; r < 8; ++r) {
        float p = __expf(s[r] - rm[r]) * rinv[r];
        arow[(size_t)(g * 8 + r) * kS + n0 + ln] = p;        // attn_dist
        pstage[wave][g * 8 + r][t * 16 + ln] = (_Float16)p;  // re-layout
      }
    }
    // Make the per-wave LDS stores visible before A-fragment loads.
    asm volatile("s_wait_dscnt 0" ::: "memory");
    v16h aP = join8(*(const v8h*)&pstage[wave][ln][g * 16],
                    *(const v8h*)&pstage[wave][ln][g * 16 + 8]);
#pragma unroll
    for (int dt = 0; dt < 4; ++dt) {
      // B 32x16: col = d (dt*16+ln), element e -> k = k0 + g*16 + e (contig).
      const _Float16* vrow = vtb + (size_t)(dt * 16 + ln) * kS + k0 + g * 16;
      acc[dt] = wmma_f16(aP, *(const v16h*)vrow, acc[dt]);
    }
  }

  // ---- Write O ----
  float* orow = outO + ((size_t)bh * kS + q0) * kD;
#pragma unroll
  for (int dt = 0; dt < 4; ++dt) {
#pragma unroll
    for (int r = 0; r < 8; ++r) {
      orow[(size_t)(g * 8 + r) * kD + dt * 16 + ln] = acc[dt][r];
    }
  }
}

extern "C" void kernel_launch(void* const* d_in, const int* in_sizes, int n_in,
                              void* d_out, int out_size, void* d_ws, size_t ws_size,
                              hipStream_t stream) {
  (void)in_sizes; (void)n_in; (void)out_size; (void)ws_size;
  const float* Q = (const float*)d_in[0];
  const float* K = (const float*)d_in[1];
  const float* V = (const float*)d_in[2];
  const unsigned char* mask = (const unsigned char*)d_in[3];

  _Float16* Kh = (_Float16*)d_ws;                       // 4 MB
  _Float16* Vt = Kh + (size_t)kBH * kS * kD;            // 4 MB

  float* outO = (float*)d_out;
  float* outA = outO + (size_t)kBH * kS * kD;

  const int n = kBH * kS * kD;                          // 2,097,152
  convert_kv<<<n / 256, 256, 0, stream>>>(K, V, Kh, Vt);
  attn_kernel<<<dim3(kS / 64, kBH), 128, 0, stream>>>(Q, Kh, Vt, mask, outO, outA);
}
